// AttentionLayerANE_26525718020514
// MI455X (gfx1250) — matrix operand order repaired and checked by reading, not compiled, not run on hardware
//
#include <hip/hip_runtime.h>
#include <hip/hip_bf16.h>
#include <stdint.h>

// ---------------------------------------------------------------------------
// AttentionLayer (RMSNorm + QKV + RoPE + GQA attn w/ sink + O-proj + residual)
// CDNA5 / gfx1250: wave32, v_wmma_f32_16x16x32_f16 for all GEMM stages,
// global_load_async_to_lds_b128 (ASYNCcnt) for attention V staging,
// f16 weights pre-converted once so GEMM inner loops are pure b128+wmma.
// ---------------------------------------------------------------------------

#define T_SEQ   2048
#define DMODEL  640
#define HEADDIM 64
#define NHEADS  14
#define NKV     2
#define HPK     7            // heads per kv group
#define QKVDIM  1152         // (14 + 2*2) * 64
#define QDIM    896          // 14 * 64
#define KVDIM   128          // 2 * 64

typedef __attribute__((ext_vector_type(16))) _Float16 v16h;
typedef __attribute__((ext_vector_type(8)))  _Float16 v8h;
typedef __attribute__((ext_vector_type(8)))  float    v8f;

static __device__ __forceinline__ v16h cat16(v8h lo, v8h hi) {
  v16h r;
#pragma unroll
  for (int i = 0; i < 8; ++i) { r[i] = lo[i]; r[i + 8] = hi[i]; }
  return r;
}

static __device__ __forceinline__ v8f wmma_f16(v16h a, v16h b, v8f c) {
  // D(16x16,f32) = A(16x32,f16) * B(32x16,f16) + C
  return __builtin_amdgcn_wmma_f32_16x16x32_f16(
      /*neg_a=*/false, a, /*neg_b=*/false, b,
      /*c_mod=*/(short)0, c, /*reuse_a=*/false, /*reuse_b=*/false);
}

// ---------------------------------------------------------------------------
// 0) one-shot f32 -> f16 weight conversion (vectorized x4)
// ---------------------------------------------------------------------------
__global__ __launch_bounds__(256) void wcvt_kernel(
    const float* __restrict__ src, _Float16* __restrict__ dst, int n4) {
  const int i = blockIdx.x * 256 + threadIdx.x;
  if (i < n4) {
    float4 f = *(const float4*)(src + i * 4);
    _Float16 h0 = (_Float16)f.x, h1 = (_Float16)f.y;
    _Float16 h2 = (_Float16)f.z, h3 = (_Float16)f.w;
    __attribute__((ext_vector_type(4))) _Float16 v = {h0, h1, h2, h3};
    *(__attribute__((ext_vector_type(4))) _Float16*)(dst + i * 4) = v;
  }
}

// ---------------------------------------------------------------------------
// 1) RMSNorm: x (1,640,1,T) f32 -> Xn [T][640] f16   (normed * w)
// ---------------------------------------------------------------------------
__global__ __launch_bounds__(256) void rmsnorm_kernel(
    const float* __restrict__ x, const float* __restrict__ w,
    _Float16* __restrict__ Xn) {
  const int t = blockIdx.x, tid = threadIdx.x;
  float v0 = x[(size_t)tid * T_SEQ + t];
  float v1 = x[(size_t)(tid + 256) * T_SEQ + t];
  float v2 = (tid < 128) ? x[(size_t)(tid + 512) * T_SEQ + t] : 0.0f;
  float ss = v0 * v0 + v1 * v1 + v2 * v2;
#pragma unroll
  for (int off = 16; off > 0; off >>= 1) ss += __shfl_down(ss, off, 32);
  __shared__ float red[8];
  __shared__ float rs_s;
  if ((tid & 31) == 0) red[tid >> 5] = ss;
  __syncthreads();
  if (tid == 0) {
    float s = 0.0f;
#pragma unroll
    for (int i = 0; i < 8; ++i) s += red[i];
    rs_s = rsqrtf(s * (1.0f / (float)DMODEL) + 1e-5f);
  }
  __syncthreads();
  const float rs = rs_s;
  Xn[(size_t)t * DMODEL + tid]       = (_Float16)(v0 * rs * w[tid]);
  Xn[(size_t)t * DMODEL + tid + 256] = (_Float16)(v1 * rs * w[tid + 256]);
  if (tid < 128)
    Xn[(size_t)t * DMODEL + tid + 512] = (_Float16)(v2 * rs * w[tid + 512]);
}

// ---------------------------------------------------------------------------
// 2) QKV GEMM: qkv[t][o] = sum_c Xn[t][c] * W[o][c] + b[o]   (f32 out)
//    One wave per 16x64 tile (1 A-frag reused by 4 B-tiles, 4 WMMA / K-step).
// ---------------------------------------------------------------------------
__global__ __launch_bounds__(32) void qkv_gemm_kernel(
    const _Float16* __restrict__ Xn, const _Float16* __restrict__ W16,
    const float* __restrict__ bias, float* __restrict__ qkvf) {
  const int lane = threadIdx.x;
  const int col = lane & 15, half = lane >> 4;
  const int tbase = blockIdx.x * 16;
  const int obase = blockIdx.y * 64;
  const _Float16* arow = Xn + (size_t)(tbase + col) * DMODEL;
  const _Float16* b0 = W16 + (size_t)(obase + col) * DMODEL;
  const _Float16* b1 = b0 + (size_t)16 * DMODEL;
  const _Float16* b2 = b0 + (size_t)32 * DMODEL;
  const _Float16* b3 = b0 + (size_t)48 * DMODEL;
  v8f acc0 = {}, acc1 = {}, acc2 = {}, acc3 = {};
#pragma unroll 2
  for (int k0 = 0; k0 < DMODEL; k0 += 32) {
    // A fragment: lane holds row M=col, K = i + 16*half (contiguous 16 halfs)
    v16h a = cat16(*(const v8h*)(arow + k0 + 16 * half),
                   *(const v8h*)(arow + k0 + 16 * half + 8));
    __builtin_prefetch(arow + k0 + 64, 0, 1);   // global_prefetch_b8
    // B fragments: lane holds col N, K = i + 16*half (contiguous 16 halfs)
    const int bo = k0 + 16 * half;
    v16h fb0 = cat16(*(const v8h*)(b0 + bo), *(const v8h*)(b0 + bo + 8));
    v16h fb1 = cat16(*(const v8h*)(b1 + bo), *(const v8h*)(b1 + bo + 8));
    v16h fb2 = cat16(*(const v8h*)(b2 + bo), *(const v8h*)(b2 + bo + 8));
    v16h fb3 = cat16(*(const v8h*)(b3 + bo), *(const v8h*)(b3 + bo + 8));
    acc0 = wmma_f16(a, fb0, acc0);
    acc1 = wmma_f16(a, fb1, acc1);
    acc2 = wmma_f16(a, fb2, acc2);
    acc3 = wmma_f16(a, fb3, acc3);
  }
#pragma unroll
  for (int j = 0; j < 4; ++j) {
    const v8f acc = (j == 0) ? acc0 : (j == 1) ? acc1 : (j == 2) ? acc2 : acc3;
    const int o = obase + j * 16 + col;
    const float bv = bias[o];
#pragma unroll
    for (int r = 0; r < 8; ++r) {
      const int t = tbase + r + 8 * half;   // C frag: M = r + 8*half
      qkvf[(size_t)t * QKVDIM + o] = acc[r] + bv;
    }
  }
}

// ---------------------------------------------------------------------------
// 3) RoPE + split + qk scale: qkvf f32 -> Qh[T][896], Kh[T][128], Vh[T][128] f16
// ---------------------------------------------------------------------------
__global__ __launch_bounds__(256) void rope_kernel(
    const float* __restrict__ qkvf, const float* __restrict__ cosb,
    const float* __restrict__ sinb, _Float16* __restrict__ Qh,
    _Float16* __restrict__ Kh, _Float16* __restrict__ Vh) {
  const int t = blockIdx.x, tid = threadIdx.x;
  const float scale = 0.35355339059327378f;   // 64^(-1/4)
  const float* row = qkvf + (size_t)t * QKVDIM;
#pragma unroll
  for (int slot = tid; slot < 512; slot += 256) {
    const int hd = slot >> 5;    // 0..13 = q heads, 14..15 = kv heads
    const int i  = slot & 31;    // half-dim index
    const float c = cosb[t * 32 + i], s = sinb[t * 32 + i];
    const float x1 = row[hd * 64 + 2 * i];
    const float x2 = row[hd * 64 + 2 * i + 1];
    const float o1 = (x1 * c - x2 * s) * scale;
    const float o2 = (x2 * c + x1 * s) * scale;
    if (hd < NHEADS) {
      Qh[(size_t)t * QDIM + hd * 64 + 2 * i]     = (_Float16)o1;
      Qh[(size_t)t * QDIM + hd * 64 + 2 * i + 1] = (_Float16)o2;
    } else {
      const int kh = hd - NHEADS;
      Kh[(size_t)t * KVDIM + kh * 64 + 2 * i]     = (_Float16)o1;
      Kh[(size_t)t * KVDIM + kh * 64 + 2 * i + 1] = (_Float16)o2;
    }
  }
  if (tid < 128)
    Vh[(size_t)t * KVDIM + tid] = (_Float16)row[1024 + tid];
}

// ---------------------------------------------------------------------------
// 4) GQA causal attention with sink, flash-style. One wave per (16 q-rows, head).
//    S^T = K x Q^T (softmax over s needs only 1 shfl_xor(16));
//    O^T = V^T x P^T with P staged in LDS, V staged via async global->LDS DMA.
// ---------------------------------------------------------------------------
__global__ __launch_bounds__(32) void attn_kernel(
    const _Float16* __restrict__ Qh, const _Float16* __restrict__ Kh,
    const _Float16* __restrict__ Vh, const float* __restrict__ sinks,
    _Float16* __restrict__ Ah) {
  __shared__ _Float16 Plds[16 * 32];   // [q][s] 1 KB
  __shared__ _Float16 Vlds[32 * 64];   // [s][d] 4 KB
  const int lane = threadIdx.x;
  const int col = lane & 15, half = lane >> 4;
  const int qbase = blockIdx.x * 16;
  const int h = blockIdx.y;
  const int kv = h / HPK;

  // Q B-fragments (N = q col, K = d): contiguous 16 halfs per lane
  const _Float16* qrow = Qh + (size_t)(qbase + col) * QDIM + h * HEADDIM;
  v16h qb[2];
#pragma unroll
  for (int kk = 0; kk < 2; ++kk) {
    v8h lo = *(const v8h*)(qrow + kk * 32 + 16 * half);
    v8h hi = *(const v8h*)(qrow + kk * 32 + 16 * half + 8);
    qb[kk] = cat16(lo, hi);
  }

  // wave-relative LDS byte address of this lane's V row (generic-ptr low 32b)
  const unsigned vlds_base = (unsigned)(uintptr_t)(&Vlds[lane * HEADDIM]);

  float m = sinks[h] * 0.69314718055994531f;   // sink logit, online-softmax init
  float l = 1.0f;                              // exp(sink - m) = 1
  v8f o[4] = {};                               // O^T accum, 4 d-tiles

  const int nchunks = (qbase >> 5) + 1;        // 32 s-values per chunk (causal)
  for (int sc = 0; sc < nchunks; ++sc) {
    const int s0 = sc * 32;
    // stage V chunk with async global->LDS DMA (lane -> one s row, 8x b128)
    {
      const _Float16* vsrc = Vh + (size_t)(s0 + lane) * KVDIM + kv * HEADDIM;
#pragma unroll
      for (int j = 0; j < HEADDIM; j += 8) {
        const unsigned la = vlds_base + (unsigned)(j * 2);
        const unsigned long long ga = (unsigned long long)(uintptr_t)(vsrc + j);
        asm volatile("global_load_async_to_lds_b128 %0, %1, off"
                     :: "v"(la), "v"(ga) : "memory");
      }
    }
    // scores S^T for two 16-s subtiles (overlaps with async V DMA)
    float sreg[2][8];
#pragma unroll
    for (int st = 0; st < 2; ++st) {
      const int sbase = s0 + st * 16;
      const _Float16* krow = Kh + (size_t)(sbase + col) * KVDIM + kv * HEADDIM;
      v8f c = {};
#pragma unroll
      for (int kk = 0; kk < 2; ++kk) {
        v8h lo = *(const v8h*)(krow + kk * 32 + 16 * half);
        v8h hi = *(const v8h*)(krow + kk * 32 + 16 * half + 8);
        c = wmma_f16(cat16(lo, hi), qb[kk], c);
      }
#pragma unroll
      for (int r = 0; r < 8; ++r) {
        const int sg = sbase + r + 8 * half;          // row of S^T = s index
        sreg[st][r] = (sg > qbase + col) ? -1.0e30f : c[r];   // causal mask
      }
    }
    // online softmax per q column (column split across lanes L and L+16)
    float tmax = -1.0e30f;
#pragma unroll
    for (int st = 0; st < 2; ++st)
#pragma unroll
      for (int r = 0; r < 8; ++r) tmax = fmaxf(tmax, sreg[st][r]);
    tmax = fmaxf(tmax, __shfl_xor(tmax, 16, 32));
    const float m_new = fmaxf(m, tmax);
    const float alpha = __expf(m - m_new);
    float psum = 0.0f;
#pragma unroll
    for (int st = 0; st < 2; ++st) {
      v8h ph;
#pragma unroll
      for (int r = 0; r < 8; ++r) {
        const float p = __expf(sreg[st][r] - m_new);
        psum += p;
        ph[r] = (_Float16)p;
      }
      // P stored [q][s] so the B-fragment read below is contiguous
      *(v8h*)&Plds[col * 32 + st * 16 + 8 * half] = ph;
    }
    psum += __shfl_xor(psum, 16, 32);
    l = l * alpha + psum;
    m = m_new;
#pragma unroll
    for (int r = 0; r < 8; ++r) {
      o[0][r] *= alpha; o[1][r] *= alpha; o[2][r] *= alpha; o[3][r] *= alpha;
    }
    // P^T B-fragment: N = q col, K = s (two 8-runs at 8h / 16+8h)
    v8h plo = *(const v8h*)&Plds[col * 32 + 8 * half];
    v8h phi = *(const v8h*)&Plds[col * 32 + 16 + 8 * half];
    v16h pb = cat16(plo, phi);
    // V DMA must have landed before the V^T gather below
    asm volatile("s_wait_asynccnt 0x0" ::: "memory");
    // O^T += V^T x P^T, one WMMA per 16-d tile
#pragma unroll
    for (int dt = 0; dt < 4; ++dt) {
      v16h va;   // A: M = d (lane&15), K = s (two 8-runs at 8h / 16+8h)
#pragma unroll
      for (int i = 0; i < 8; ++i) {
        va[i]     = Vlds[(8 * half + i) * HEADDIM + dt * 16 + col];
        va[i + 8] = Vlds[(16 + 8 * half + i) * HEADDIM + dt * 16 + col];
      }
      o[dt] = wmma_f16(va, pb, o[dt]);
    }
  }
  // normalize + store attn_out as f16 [t][h*64+d] (vectorized 16B stores)
  const float inv_l = 1.0f / l;
  _Float16* arow = Ah + (size_t)(qbase + col) * QDIM + h * HEADDIM;
#pragma unroll
  for (int dt = 0; dt < 4; ++dt) {
    v8h w;
#pragma unroll
    for (int r = 0; r < 8; ++r) w[r] = (_Float16)(o[dt][r] * inv_l);
    *(v8h*)(arow + dt * 16 + 8 * half) = w;
  }
}

// ---------------------------------------------------------------------------
// 5) O-proj + bias + residual: out[d][t] = x[d][t] + sum_o Ah[t][o]*W[d][o] + b[d]
//    One wave per 16x64 tile (4 B-tiles share the A fragment).
// ---------------------------------------------------------------------------
__global__ __launch_bounds__(32) void oproj_kernel(
    const _Float16* __restrict__ Ah, const _Float16* __restrict__ W16,
    const float* __restrict__ ob, const float* __restrict__ x,
    float* __restrict__ out) {
  const int lane = threadIdx.x;
  const int col = lane & 15, half = lane >> 4;
  const int tbase = blockIdx.x * 16;
  const int dbase = blockIdx.y * 64;
  const _Float16* arow = Ah + (size_t)(tbase + col) * QDIM;
  const _Float16* b0 = W16 + (size_t)(dbase + col) * QDIM;
  const _Float16* b1 = b0 + (size_t)16 * QDIM;
  const _Float16* b2 = b0 + (size_t)32 * QDIM;
  const _Float16* b3 = b0 + (size_t)48 * QDIM;
  v8f acc0 = {}, acc1 = {}, acc2 = {}, acc3 = {};
#pragma unroll 2
  for (int k0 = 0; k0 < QDIM; k0 += 32) {
    v16h a = cat16(*(const v8h*)(arow + k0 + 16 * half),
                   *(const v8h*)(arow + k0 + 16 * half + 8));
    const int bo = k0 + 16 * half;
    v16h fb0 = cat16(*(const v8h*)(b0 + bo), *(const v8h*)(b0 + bo + 8));
    v16h fb1 = cat16(*(const v8h*)(b1 + bo), *(const v8h*)(b1 + bo + 8));
    v16h fb2 = cat16(*(const v8h*)(b2 + bo), *(const v8h*)(b2 + bo + 8));
    v16h fb3 = cat16(*(const v8h*)(b3 + bo), *(const v8h*)(b3 + bo + 8));
    acc0 = wmma_f16(a, fb0, acc0);
    acc1 = wmma_f16(a, fb1, acc1);
    acc2 = wmma_f16(a, fb2, acc2);
    acc3 = wmma_f16(a, fb3, acc3);
  }
  // epilogue per tile: 8 consecutive t's per lane -> two float4 loads/stores
#pragma unroll
  for (int j = 0; j < 4; ++j) {
    const v8f acc = (j == 0) ? acc0 : (j == 1) ? acc1 : (j == 2) ? acc2 : acc3;
    const int d = dbase + j * 16 + col;
    const float bv = ob[d];
    const float* xr = x + (size_t)d * T_SEQ + tbase + 8 * half;
    float* orow = out + (size_t)d * T_SEQ + tbase + 8 * half;
    float4 x0 = *(const float4*)(xr);
    float4 x1 = *(const float4*)(xr + 4);
    float4 w0, w1;
    w0.x = x0.x + acc[0] + bv; w0.y = x0.y + acc[1] + bv;
    w0.z = x0.z + acc[2] + bv; w0.w = x0.w + acc[3] + bv;
    w1.x = x1.x + acc[4] + bv; w1.y = x1.y + acc[5] + bv;
    w1.z = x1.z + acc[6] + bv; w1.w = x1.w + acc[7] + bv;
    *(float4*)(orow) = w0;
    *(float4*)(orow + 4) = w1;
  }
}

// ---------------------------------------------------------------------------
extern "C" void kernel_launch(void* const* d_in, const int* in_sizes, int n_in,
                              void* d_out, int out_size, void* d_ws, size_t ws_size,
                              hipStream_t stream) {
  (void)in_sizes; (void)n_in; (void)out_size; (void)ws_size;
  const float* x        = (const float*)d_in[0];
  const float* rope_cos = (const float*)d_in[1];
  const float* rope_sin = (const float*)d_in[2];
  // d_in[3] = attn_mask (causal mask applied analytically in attn_kernel)
  const float* norm_w   = (const float*)d_in[4];
  const float* qkv_w    = (const float*)d_in[5];
  const float* qkv_b    = (const float*)d_in[6];
  const float* o_w      = (const float*)d_in[7];
  const float* o_b      = (const float*)d_in[8];
  const float* sinks    = (const float*)d_in[9];
  float* out = (float*)d_out;

  char* ws = (char*)d_ws;
  auto alignup = [](size_t v) { return (v + 255) & ~(size_t)255; };
  size_t off = 0;
  _Float16* Xn   = (_Float16*)(ws + off); off += alignup((size_t)T_SEQ * DMODEL * 2);
  float*    qkvf = (float*)(ws + off);    off += alignup((size_t)T_SEQ * QKVDIM * 4);
  _Float16* Qh   = (_Float16*)(ws + off); off += alignup((size_t)T_SEQ * QDIM * 2);
  _Float16* Kh   = (_Float16*)(ws + off); off += alignup((size_t)T_SEQ * KVDIM * 2);
  _Float16* Vh   = (_Float16*)(ws + off); off += alignup((size_t)T_SEQ * KVDIM * 2);
  _Float16* Ah   = (_Float16*)(ws + off); off += alignup((size_t)T_SEQ * QDIM * 2);
  _Float16* Wq16 = (_Float16*)(ws + off); off += alignup((size_t)QKVDIM * DMODEL * 2);
  _Float16* Wo16 = (_Float16*)(ws + off); off += alignup((size_t)DMODEL * QDIM * 2);

  const int nq4 = (QKVDIM * DMODEL) / 4;   // 184320
  const int no4 = (DMODEL * QDIM) / 4;     // 143360
  wcvt_kernel<<<(nq4 + 255) / 256, 256, 0, stream>>>(qkv_w, Wq16, nq4);
  wcvt_kernel<<<(no4 + 255) / 256, 256, 0, stream>>>(o_w, Wo16, no4);

  rmsnorm_kernel<<<T_SEQ, 256, 0, stream>>>(x, norm_w, Xn);
  qkv_gemm_kernel<<<dim3(T_SEQ / 16, QKVDIM / 64), 32, 0, stream>>>(Xn, Wq16, qkv_b, qkvf);
  rope_kernel<<<T_SEQ, 256, 0, stream>>>(qkvf, rope_cos, rope_sin, Qh, Kh, Vh);
  attn_kernel<<<dim3(T_SEQ / 16, NHEADS), 32, 0, stream>>>(Qh, Kh, Vh, sinks, Ah);
  oproj_kernel<<<dim3(T_SEQ / 16, DMODEL / 64), 32, 0, stream>>>(Ah, Wo16, o_b, x, out);
}
